// VMRNN_10840497455220
// MI455X (gfx1250) — matrix operand order repaired and checked by reading, not compiled, not run on hardware
//
#include <hip/hip_runtime.h>
#include <math.h>

// Problem dims
#define B_   256
#define T_   512
#define DM_  256
#define H_   256
#define NL_  3
#define DI_  512
#define DS_  16
#define DTR_ 16

// Padded LDS row strides (in bf16 elements): +8 elems = +16B pad -> conflict-free b128 loads
#define STRA 264   // for 256-wide rows
#define STRB 520   // for 512-wide rows

typedef float  v8f   __attribute__((ext_vector_type(8)));
typedef __bf16 v16bf __attribute__((ext_vector_type(16)));

union V16u {
    v16bf  v;
    uint4  q[2];
    __bf16 e[16];
};

// ---- static device buffers: bf16 copies of x and GEMM weights (all (N,K) row-major) ----
__device__ __align__(16) __bf16 g_x[(size_t)B_ * T_ * DM_];       // 33.5M elems
__device__ __align__(16) __bf16 g_w_in[NL_ * 2 * DI_ * DM_];      // in_proj  (1024,256)
__device__ __align__(16) __bf16 g_w_xp[NL_ * (DTR_ + 2 * DS_) * DI_]; // x_proj (48,512)
__device__ __align__(16) __bf16 g_w_dt[NL_ * DI_ * DTR_];         // dt_w     (512,16)
__device__ __align__(16) __bf16 g_w_op[NL_ * DM_ * DI_];          // out_proj (256,512)
__device__ __align__(16) __bf16 g_w_ih[NL_ * 4 * H_ * DM_];       // w_ih     (1024,256)
__device__ __align__(16) __bf16 g_w_hh[NL_ * 4 * H_ * H_];        // w_hh     (1024,256)

// ---- fp32 -> bf16 conversion into the device globals ----
__global__ void cvt_kernel(const float* __restrict__ src, int n, int which) {
    __bf16* dst;
    switch (which) {
        case 0: dst = g_x;    break;
        case 1: dst = g_w_in; break;
        case 2: dst = g_w_xp; break;
        case 3: dst = g_w_dt; break;
        case 4: dst = g_w_op; break;
        case 5: dst = g_w_ih; break;
        default: dst = g_w_hh; break;
    }
    for (int i = blockIdx.x * blockDim.x + threadIdx.x; i < n; i += gridDim.x * blockDim.x)
        dst[i] = (__bf16)src[i];
}

// ---- helpers ----
static __device__ __forceinline__ v8f zero8() {
    v8f z = {0.f, 0.f, 0.f, 0.f, 0.f, 0.f, 0.f, 0.f};
    return z;
}
static __device__ __forceinline__ float sigmoidf_(float x) { return 1.f / (1.f + __expf(-x)); }
static __device__ __forceinline__ float siluf_(float x)    { return x * sigmoidf_(x); }
static __device__ __forceinline__ float softplusf_(float x){ return (x > 20.f) ? x : logf(1.f + __expf(x)); }

static __device__ __forceinline__ v8f wmma_bf16(v16bf a, v16bf b, v8f c) {
    return __builtin_amdgcn_wmma_f32_16x16x32_bf16(false, a, false, b, (short)0, c, false, false);
}

// A operand (16x32, 16-bit): lane = row m (mod 16); lanes<16 hold K{0..7,16..23}, lanes>=16 hold K{8..15,24..31}
static __device__ __forceinline__ v16bf load_a16(const __bf16* base, int strideE, int kt, int lane) {
    int m  = lane & 15;
    int k0 = kt * 32 + ((lane >> 4) << 3);
    const __bf16* p = base + m * strideE + k0;
    V16u r;
    r.q[0] = *(const uint4*)p;
    r.q[1] = *(const uint4*)(p + 16);
    return r.v;
}

// B operand (32x16, 16-bit) from (N,K) row-major weights: lane = col n; lanes<16 hold K{0..7,16..23}, lanes>=16 hold K{8..15,24..31}
static __device__ __forceinline__ v16bf load_b16(const __bf16* wmat, int K, int n0, int kt, int lane) {
    int n  = n0 + (lane & 15);
    int k0 = kt * 32 + ((lane >> 4) << 3);
    const __bf16* p = wmat + (size_t)n * K + k0;
    V16u r;
    r.q[0] = *(const uint4*)p;
    r.q[1] = *(const uint4*)(p + 16);
    return r.v;
}

// ---- main persistent kernel: grid = 16 WGs x 16 batch rows, 512 threads (16 waves) each ----
__global__ __launch_bounds__(512) void vmrnn_main(
    const float* __restrict__ conv_w, const float* __restrict__ conv_b,
    const float* __restrict__ dt_b,   const float* __restrict__ ssm_D,
    const float* __restrict__ b_ih,   const float* __restrict__ b_hh,
    const float* __restrict__ fc_w,   const float* __restrict__ fc_b,
    float* __restrict__ out)
{
    __shared__ __align__(16) __bf16 sX [16 * STRA];        // layer-0 input rows (bf16)
    __shared__ __align__(16) __bf16 sH [NL_ * 16 * STRA];  // per-layer h state (bf16)
    __shared__ __align__(16) __bf16 sXc[16 * STRB];
    __shared__ __align__(16) __bf16 sZs[16 * STRB];
    __shared__ __align__(16) __bf16 sY [16 * STRB];
    __shared__ __align__(16) __bf16 sM [16 * STRA];
    __shared__ __align__(16) float  sDbl[16 * 48];
    __shared__            float  sBC[16];

    const int tid  = threadIdx.x;
    const int lane = tid & 31;
    const int w    = tid >> 5;          // wave id 0..15
    const int ln16 = lane & 15;
    const int mrow = (lane >> 4) * 8;   // C/D tile: m = v + mrow
    const int r0   = blockIdx.x * 16;   // first batch row owned by this WG

    for (int idx = tid; idx < NL_ * 16 * STRA; idx += 512) sH[idx] = (__bf16)0.f;
    float c_reg[NL_][8];
    #pragma unroll
    for (int i = 0; i < NL_; ++i)
        #pragma unroll
        for (int v = 0; v < 8; ++v) c_reg[i][v] = 0.f;
    const float fcb = fc_b[0];
    __syncthreads();

    for (int t = 0; t < T_; ++t) {
        // stage x[:, t, :] rows for this WG into LDS (8 bf16 per thread)
        {
            int idx = tid * 8, row = idx >> 8, col = idx & 255;
            const uint4* src = (const uint4*)(g_x + ((size_t)(r0 + row) * T_ + t) * DM_ + col);
            *(uint4*)(sX + row * STRA + col) = *src;
        }
        __syncthreads();

        #pragma unroll
        for (int i = 0; i < NL_; ++i) {
            const __bf16* inp = (i == 0) ? sX : (sH + (i - 1) * 16 * STRA);
            const int strIn = STRA;

            // ---- phase 1: xz = inp @ in_proj^T  (N=1024, K=256), fused conv-tap+silu / silu(z)
            {
                const __bf16* Win = g_w_in + (size_t)i * 1024 * DM_;
                v8f acc[4];
                #pragma unroll
                for (int j = 0; j < 4; ++j) acc[j] = zero8();
                #pragma unroll
                for (int kt = 0; kt < 8; ++kt) {
                    v16bf A = load_a16(inp, strIn, kt, lane);
                    #pragma unroll
                    for (int j = 0; j < 4; ++j) {
                        int nt = w + 16 * j;
                        acc[j] = wmma_bf16(A, load_b16(Win, DM_, nt * 16, kt, lane), acc[j]);
                    }
                }
                #pragma unroll
                for (int j = 0; j < 4; ++j) {
                    int nt = w + 16 * j;
                    int col = nt * 16 + ln16;
                    if (nt < 32) {
                        float cw = conv_w[((size_t)i * DI_ + col) * 4 + 3];
                        float cb = conv_b[i * DI_ + col];
                        #pragma unroll
                        for (int v = 0; v < 8; ++v) {
                            int m = v + mrow;
                            sXc[m * STRB + col] = (__bf16)siluf_(acc[j][v] * cw + cb);
                        }
                    } else {
                        int c2 = col - 512;
                        #pragma unroll
                        for (int v = 0; v < 8; ++v) {
                            int m = v + mrow;
                            sZs[m * STRB + c2] = (__bf16)siluf_(acc[j][v]);
                        }
                    }
                }
            }
            __syncthreads();

            // ---- phase 2: dbl = xc @ x_proj^T  (N=48, K=512), fp32 out to LDS
            if (w < 3) {
                const __bf16* Wxp = g_w_xp + (size_t)i * 48 * DI_;
                v8f a2 = zero8();
                #pragma unroll
                for (int kt = 0; kt < 16; ++kt)
                    a2 = wmma_bf16(load_a16(sXc, STRB, kt, lane),
                                   load_b16(Wxp, DI_, w * 16, kt, lane), a2);
                int n = w * 16 + ln16;
                #pragma unroll
                for (int v = 0; v < 8; ++v) sDbl[(v + mrow) * 48 + n] = a2[v];
            }
            __syncthreads();

            // ---- phase 3 (wave 3): bc[m] = sum_s Bs*Cs ; phase 4 GEMM: delta-pre = dt @ dt_w^T (N=512, K=16 padded)
            if (w == 3 && lane < 16) {
                float s = 0.f;
                #pragma unroll
                for (int q = 0; q < 16; ++q)
                    s += sDbl[lane * 48 + 16 + q] * sDbl[lane * 48 + 32 + q];
                sBC[lane] = s;
            }
            v8f accd[2];
            #pragma unroll
            for (int j = 0; j < 2; ++j) {
                int nt = w * 2 + j;
                V16u a;
                int k0a = (lane >> 4) << 3;   // 0 or 8 (real dt cols); K>=16 is zero pad
                #pragma unroll
                for (int q = 0; q < 8; ++q) {
                    a.e[q]     = (__bf16)sDbl[ln16 * 48 + k0a + q];
                    a.e[8 + q] = (__bf16)0.f;
                }
                V16u b;
                {
                    const __bf16* p = g_w_dt + ((size_t)i * DI_ + nt * 16 + ln16) * DTR_;
                    b.q[0] = ((const uint4*)p)[lane >> 4];
                    b.q[1] = make_uint4(0u, 0u, 0u, 0u);   // K 16..31 = zero pad
                }
                accd[j] = wmma_bf16(a.v, b.v, zero8());
            }
            __syncthreads();   // sBC ready
            #pragma unroll
            for (int j = 0; j < 2; ++j) {
                int nt  = w * 2 + j;
                int col = nt * 16 + ln16;
                float db = dt_b[i * DI_ + col];
                float Dv = ssm_D[i * DI_ + col];
                #pragma unroll
                for (int v = 0; v < 8; ++v) {
                    int m = v + mrow;
                    float delta = softplusf_(accd[j][v] + db);
                    float xcv = (float)sXc[m * STRB + col];
                    float zv  = (float)sZs[m * STRB + col];
                    sY[m * STRB + col] = (__bf16)((delta * sBC[m] + Dv) * xcv * zv);
                }
            }
            __syncthreads();

            // ---- phase 5: m_out = y @ out_proj^T  (N=256, K=512)
            {
                const __bf16* Wop = g_w_op + (size_t)i * DM_ * DI_;
                v8f a5 = zero8();
                #pragma unroll
                for (int kt = 0; kt < 16; ++kt)
                    a5 = wmma_bf16(load_a16(sY, STRB, kt, lane),
                                   load_b16(Wop, DI_, w * 16, kt, lane), a5);
                int col = w * 16 + ln16;
                #pragma unroll
                for (int v = 0; v < 8; ++v) sM[(v + mrow) * STRA + col] = (__bf16)a5[v];
            }
            __syncthreads();

            // ---- phase 6: gates = m_out @ w_ih^T + h_old @ w_hh^T; fused LSTM cell (c in regs)
            {
                const __bf16* Wih = g_w_ih + (size_t)i * 1024 * DM_;
                const __bf16* Whh = g_w_hh + (size_t)i * 1024 * H_;
                const __bf16* hOld = sH + i * 16 * STRA;
                v8f g[4];
                #pragma unroll
                for (int q = 0; q < 4; ++q) g[q] = zero8();
                #pragma unroll
                for (int kt = 0; kt < 8; ++kt) {
                    v16bf A = load_a16(sM, STRA, kt, lane);
                    #pragma unroll
                    for (int q = 0; q < 4; ++q)
                        g[q] = wmma_bf16(A, load_b16(Wih, DM_, q * 256 + w * 16, kt, lane), g[q]);
                }
                #pragma unroll
                for (int kt = 0; kt < 8; ++kt) {
                    v16bf A = load_a16(hOld, STRA, kt, lane);
                    #pragma unroll
                    for (int q = 0; q < 4; ++q)
                        g[q] = wmma_bf16(A, load_b16(Whh, H_, q * 256 + w * 16, kt, lane), g[q]);
                }
                __syncthreads();   // all reads of old h complete before overwrite
                int colh = w * 16 + ln16;
                float bs[4];
                #pragma unroll
                for (int q = 0; q < 4; ++q)
                    bs[q] = b_ih[i * 1024 + q * 256 + colh] + b_hh[i * 1024 + q * 256 + colh];
                __bf16* hNew = sH + i * 16 * STRA;
                #pragma unroll
                for (int v = 0; v < 8; ++v) {
                    int m = v + mrow;
                    float ig = g[0][v] + bs[0];
                    float fg = g[1][v] + bs[1];
                    float gg = g[2][v] + bs[2];
                    float og = g[3][v] + bs[3];
                    float cn = sigmoidf_(fg) * c_reg[i][v] + sigmoidf_(ig) * tanhf(gg);
                    float hn = sigmoidf_(og) * tanhf(cn);
                    c_reg[i][v] = cn;
                    hNew[m * STRA + colh] = (__bf16)hn;
                }
            }
            __syncthreads();
        } // layers

        // ---- fused fc: out[b,t] = h2 . fc_w + fc_b ; one wave per local row
        {
            const __bf16* h2 = sH + 2 * 16 * STRA;
            int m = w;
            float s = 0.f;
            #pragma unroll
            for (int jj = 0; jj < 8; ++jj) {
                int k = lane + 32 * jj;
                s += (float)h2[m * STRA + k] * fc_w[k];
            }
            #pragma unroll
            for (int off = 16; off >= 1; off >>= 1) s += __shfl_xor(s, off);
            if (lane == 0) out[(size_t)(r0 + m) * T_ + t] = s + fcb;
        }
        __syncthreads();
    }
}

extern "C" void kernel_launch(void* const* d_in, const int* in_sizes, int n_in,
                              void* d_out, int out_size, void* d_ws, size_t ws_size,
                              hipStream_t stream) {
    (void)in_sizes; (void)n_in; (void)out_size; (void)d_ws; (void)ws_size;

    auto cvt = [&](const void* src, int n, int which) {
        int blocks = (n + 255) / 256;
        if (blocks > 2048) blocks = 2048;
        cvt_kernel<<<blocks, 256, 0, stream>>>((const float*)src, n, which);
    };
    cvt(d_in[0],  B_ * T_ * DM_,            0);  // x
    cvt(d_in[1],  NL_ * 2 * DI_ * DM_,      1);  // in_proj
    cvt(d_in[4],  NL_ * (DTR_ + 2*DS_) * DI_, 2); // x_proj
    cvt(d_in[5],  NL_ * DI_ * DTR_,         3);  // dt_w
    cvt(d_in[8],  NL_ * DM_ * DI_,          4);  // out_proj
    cvt(d_in[9],  NL_ * 4 * H_ * DM_,       5);  // w_ih
    cvt(d_in[10], NL_ * 4 * H_ * H_,        6);  // w_hh

    vmrnn_main<<<B_ / 16, 512, 0, stream>>>(
        (const float*)d_in[2],  (const float*)d_in[3],   // conv_w, conv_b
        (const float*)d_in[6],  (const float*)d_in[7],   // dt_b, ssm_D
        (const float*)d_in[11], (const float*)d_in[12],  // b_ih, b_hh
        (const float*)d_in[13], (const float*)d_in[14],  // fc_w, fc_b
        (float*)d_out);
}
